// LoFTREncoderLayer_31602369364048
// MI455X (gfx1250) — matrix operand order repaired and checked by reading, not compiled, not run on hardware
//
#include <hip/hip_runtime.h>

// ---------------------------------------------------------------------------
// LoFTR encoder layer for MI455X (gfx1250).
// All large GEMMs run as bf16 WMMA (v_wmma_f32_16x16x32_bf16, f32 accum).
// Block tile 64(M)x32(N): 4 waves x (16x32 per wave, 2 WMMAs per A fragment).
// The shared B tile is staged through LDS, using gfx1250 async-to-LDS loads
// (ASYNCcnt) when the toolchain exposes them, else plain LDS stores.
// Reductions / softmax / layernorm stay f32 VALU. Weights are transposed to
// [N,K] bf16 once so every WMMA fragment load is contiguous per lane.
// ---------------------------------------------------------------------------

typedef __bf16 bf16_t;
typedef __attribute__((ext_vector_type(16))) __bf16 v16bf;
typedef __attribute__((ext_vector_type(8)))  float  v8f;
typedef int v4i_gcc __attribute__((vector_size(16)));

#define B_  8
#define L_  4800
#define S_  4800
#define C_  256
#define H_  8
#define NP_ 64
#define BL_ (B_ * L_)

#if defined(__has_builtin)
#if __has_builtin(__builtin_amdgcn_global_load_async_to_lds_b128) && \
    __has_builtin(__builtin_amdgcn_s_wait_asynccnt)
#define HAVE_ASYNC_LDS 1
#endif
#endif
#ifndef HAVE_ASYNC_LDS
#define HAVE_ASYNC_LDS 0
#endif

__device__ __forceinline__ float elu1(float x) { return x > 0.f ? x + 1.f : expf(x); }

// ---------------------------------------------------------------------------
// Generic batched GEMM:  C[M,N] = A[M,K] * Bt[N,K]^T   (A,Bt bf16 row-major)
// M % 64 == 0 (or grid sized exactly), N % 32 == 0, K % 32 == 0.
// ---------------------------------------------------------------------------
__device__ __forceinline__ v16bf load_a_frag(const bf16_t* __restrict__ A, int ld,
                                             int m0, int k0, int lane) {
  // 16-bit A 16x32 layout: lanes 0-15 rows M=0..15 K={0..7,16..23};
  // lanes 16-31 same rows, K={8..15,24..31}.
  const bf16_t* p = A + (long)(m0 + (lane & 15)) * ld + (k0 + ((lane & 16) ? 8 : 0));
  v16bf a;
#pragma unroll
  for (int i = 0; i < 8; ++i) { a[i] = p[i]; a[i + 8] = p[i + 16]; }
  return a;
}

template <int MODE>  // 0: f32 out, 1: bf16 out, 2: bf16 relu out
__global__ void gemm_nt(const bf16_t* __restrict__ A, const bf16_t* __restrict__ Bt,
                        void* __restrict__ Out, int M, int N, int K,
                        long sA, long sB, long sC) {
  __shared__ bf16_t tileB[32][32];                   // 32 cols x 32 k (2 KB)
  const int lane = threadIdx.x;                      // 0..31
  const int tid  = threadIdx.y * 32 + lane;          // 0..127
  const int m0 = blockIdx.y * 64 + threadIdx.y * 16; // 4 waves stacked in M
  const int n0 = blockIdx.x * 32;
  const bf16_t* Ab = A + (long)blockIdx.z * sA;
  const bf16_t* Bb = Bt + (long)blockIdx.z * sB;
  const int brow = tid >> 2;                         // 0..31  (col within tile)
  const int bchk = (tid & 3) * 8;                    // 0,8,16,24 (k within tile)
  v8f acc0 = {}, acc1 = {};
  for (int k0 = 0; k0 < K; k0 += 32) {
    // ---- stage the shared B tile into LDS (one 16B chunk per thread) ----
    const bf16_t* gB = Bb + (long)(n0 + brow) * K + (k0 + bchk);
#if HAVE_ASYNC_LDS
    __builtin_amdgcn_global_load_async_to_lds_b128(
        (__attribute__((address_space(1))) v4i_gcc*)gB,
        (__attribute__((address_space(3))) v4i_gcc*)&tileB[brow][bchk], 0, 0);
    __builtin_amdgcn_s_wait_asynccnt(0);   // wave waits its own async ops
#else
#pragma unroll
    for (int i = 0; i < 8; ++i) tileB[brow][bchk + i] = gB[i];
#endif
    __syncthreads();
    if (k0 + 32 < K) {  // gfx1250 global_prefetch_b8 of next K tiles
      __builtin_prefetch(Ab + (long)(m0 + (lane & 15)) * K + (k0 + 32), 0, 3);
      __builtin_prefetch(Bb + (long)(n0 + brow) * K + (k0 + 32), 0, 3);
    }
    v16bf af = load_a_frag(Ab, K, m0, k0, lane);
    // B frag: lanes 0-15 col N, K={0..7,16..23}; lanes 16-31 K={8..15,24..31}.
    const int koff = (lane & 16) ? 8 : 0;
    const int ncol = lane & 15;
    v16bf b0, b1;
#pragma unroll
    for (int i = 0; i < 8; ++i) {
      b0[i] = tileB[ncol][koff + i];
      b0[i + 8] = tileB[ncol][koff + 16 + i];
      b1[i] = tileB[16 + ncol][koff + i];
      b1[i + 8] = tileB[16 + ncol][koff + 16 + i];
    }
    acc0 = __builtin_amdgcn_wmma_f32_16x16x32_bf16(false, af, false, b0,
                                                   (short)0, acc0, false, false);
    acc1 = __builtin_amdgcn_wmma_f32_16x16x32_bf16(false, af, false, b1,
                                                   (short)0, acc1, false, false);
    __syncthreads();
  }
  // C/D layout: lane l -> N = n0 + (l&15); VGPR r -> M = m0 + r + 8*(l>=16)
  const int mb = m0 + ((lane & 16) ? 8 : 0);
  long base0 = (long)blockIdx.z * sC + (long)mb * N + (n0 + (lane & 15));
  long base1 = base0 + 16;
#pragma unroll
  for (int r = 0; r < 8; ++r) {
    float v0 = acc0[r], v1 = acc1[r];
    long i0 = base0 + (long)r * N, i1 = base1 + (long)r * N;
    if (MODE == 0) {
      ((float*)Out)[i0] = v0; ((float*)Out)[i1] = v1;
    } else if (MODE == 1) {
      ((bf16_t*)Out)[i0] = (bf16_t)v0; ((bf16_t*)Out)[i1] = (bf16_t)v1;
    } else {
      ((bf16_t*)Out)[i0] = (bf16_t)(v0 > 0.f ? v0 : 0.f);
      ((bf16_t*)Out)[i1] = (bf16_t)(v1 > 0.f ? v1 : 0.f);
    }
  }
}

// ---------------------------------------------------------------------------
// Elementwise helpers
// ---------------------------------------------------------------------------
__global__ void conv_f2b_kernel(const float* __restrict__ in, bf16_t* __restrict__ out, long n) {
  long i = (long)blockIdx.x * 256 + threadIdx.x;
  if (i < n) out[i] = (bf16_t)in[i];
}

// W [K,N] f32 row-major  ->  Wt [N,K] bf16 row-major
__global__ void wconv_kernel(const float* __restrict__ W, bf16_t* __restrict__ Wt, int K, int N) {
  long i = (long)blockIdx.x * 256 + threadIdx.x;
  if (i >= (long)K * N) return;
  int k = (int)(i / N), n = (int)(i % N);
  Wt[(long)n * K + k] = (bf16_t)W[i];
}

__global__ void concat_kernel(const bf16_t* __restrict__ xb, const bf16_t* __restrict__ mb,
                              bf16_t* __restrict__ out, long rows) {
  long i = (long)blockIdx.x * 256 + threadIdx.x;
  if (i >= rows * 512) return;
  long r = i >> 9; int c = (int)(i & 511);
  out[i] = (c < 256) ? xb[r * 256 + c] : mb[r * 256 + (c - 256)];
}

// ---------------------------------------------------------------------------
// LayerNorm over C=256 with optional residual (resid_mod: broadcast modulo),
// optional post-add (out = LN(in+res)*g+b + post), f32 and/or bf16 outputs.
// ---------------------------------------------------------------------------
__global__ void ln_kernel(const float* __restrict__ in, const float* __restrict__ resid,
                          int resid_mod, const float* __restrict__ postadd,
                          const float* __restrict__ g, const float* __restrict__ bta,
                          float* __restrict__ outf, bf16_t* __restrict__ outb) {
  long row = blockIdx.x;
  int c = threadIdx.x;
  float v = in[row * 256 + c];
  if (resid) {
    long rr = resid_mod ? (row % resid_mod) : row;
    v += resid[rr * 256 + c];
  }
  __shared__ float red[256];
  red[c] = v; __syncthreads();
  for (int s = 128; s > 0; s >>= 1) { if (c < s) red[c] += red[c + s]; __syncthreads(); }
  float mean = red[0] * (1.f / 256.f);
  __syncthreads();
  float d = v - mean;
  red[c] = d * d; __syncthreads();
  for (int s = 128; s > 0; s >>= 1) { if (c < s) red[c] += red[c + s]; __syncthreads(); }
  float var = red[0] * (1.f / 256.f);
  float y = d * rsqrtf(var + 1e-5f) * g[c] + bta[c];
  if (postadd) y += postadd[row * 256 + c];
  if (outf) outf[row * 256 + c] = y;
  if (outb) outb[row * 256 + c] = (bf16_t)y;
}

// ---------------------------------------------------------------------------
// Decoder self-attention (batch-independent, length 64, H=8, DH=32).
// ---------------------------------------------------------------------------
__global__ void kv1_kernel(const float* __restrict__ qk, const float* __restrict__ qv,
                           float* __restrict__ KV, float* __restrict__ KS) {
  int h = blockIdx.x, t = threadIdx.x;
  int v = t & 31, db = t >> 5;  // db 0..7, 4 d's per thread
  float acc[4] = {0.f, 0.f, 0.f, 0.f}, ks = 0.f;
  for (int s = 0; s < 64; ++s) {
    float vv = qv[s * 256 + h * 32 + v];
#pragma unroll
    for (int j = 0; j < 4; ++j)
      acc[j] += elu1(qk[s * 256 + h * 32 + db + 8 * j]) * vv;
    if (t < 32) ks += elu1(qk[s * 256 + h * 32 + t]);
  }
#pragma unroll
  for (int j = 0; j < 4; ++j) KV[h * 1024 + (db + 8 * j) * 32 + v] = acc[j];
  if (t < 32) KS[h * 32 + t] = ks;
}

__global__ void attn_apply1(const float* __restrict__ qq, const float* __restrict__ KV,
                            const float* __restrict__ KS, float* __restrict__ outf) {
  int l = blockIdx.x, t = threadIdx.x, h = t >> 5, v = t & 31;
  const float* qrow = qq + l * 256 + h * 32;
  const float* ks = KS + h * 32;
  const float* kv = KV + h * 1024;
  float z = 0.f, acc = 0.f;
#pragma unroll 8
  for (int d = 0; d < 32; ++d) {
    float qe = elu1(qrow[d]);
    z += qe * ks[d];
    acc += qe * kv[d * 32 + v];
  }
  outf[l * 256 + t] = acc / (z + 1e-6f);
}

// ---------------------------------------------------------------------------
// Decoder cross-attention KV reduction over L=4800 (fk,fv bf16, masked).
// ---------------------------------------------------------------------------
__global__ void kv2_kernel(const bf16_t* __restrict__ fk, const bf16_t* __restrict__ fv,
                           const unsigned char* __restrict__ mask,
                           float* __restrict__ KV, float* __restrict__ KS, int Lr) {
  int bh = blockIdx.x, b = bh >> 3, h = bh & 7;
  int t = threadIdx.x, v = t & 31, db = t >> 5;
  __shared__ float sk[4][32], sv[4][32];
  float acc[4] = {0.f, 0.f, 0.f, 0.f}, ks = 0.f;
  const bf16_t* fkb = fk + (long)b * Lr * 256;
  const bf16_t* fvb = fv + (long)b * Lr * 256;
  const unsigned char* mb = mask + (long)b * Lr;
  for (int s0 = 0; s0 < Lr; s0 += 4) {
    if (t < 128) {
      int i = t >> 5, d = t & 31;
      float mm = mb[s0 + i] ? 1.f : 0.f;
      sk[i][d] = elu1((float)fkb[(long)(s0 + i) * 256 + h * 32 + d]) * mm;
    } else {
      int u = t - 128, i = u >> 5, vv = u & 31;
      float mm = mb[s0 + i] ? 1.f : 0.f;
      sv[i][vv] = (float)fvb[(long)(s0 + i) * 256 + h * 32 + vv] * mm;
    }
    __syncthreads();
#pragma unroll
    for (int i = 0; i < 4; ++i) {
      float vv = sv[i][v];
#pragma unroll
      for (int j = 0; j < 4; ++j) acc[j] += sk[i][db + 8 * j] * vv;
      if (t < 32) ks += sk[i][t];
    }
    __syncthreads();
  }
#pragma unroll
  for (int j = 0; j < 4; ++j) KV[(long)bh * 1024 + (db + 8 * j) * 32 + v] = acc[j];
  if (t < 32) KS[bh * 32 + t] = ks;
}

__global__ void attn_apply2(const float* __restrict__ mq, const float* __restrict__ KV,
                            const float* __restrict__ KS, bf16_t* __restrict__ outb) {
  int l = blockIdx.x, b = blockIdx.y;
  int t = threadIdx.x, h = t >> 5, v = t & 31;
  const float* qrow = mq + l * 256 + h * 32;
  const float* ks = KS + (b * 8 + h) * 32;
  const float* kv = KV + (long)(b * 8 + h) * 1024;
  float z = 0.f, acc = 0.f;
#pragma unroll 8
  for (int d = 0; d < 32; ++d) {
    float qe = elu1(qrow[d]);
    z += qe * ks[d];
    acc += qe * kv[d * 32 + v];
  }
  outb[(long)(b * 64 + l) * 256 + t] = (bf16_t)(acc / (z + 1e-6f));
}

// ---------------------------------------------------------------------------
// Double softmax: q[B,L,64] -> softmax(axis=1) * softmax(axis=2), masked.
// ---------------------------------------------------------------------------
__global__ void colstats_kernel(const float* __restrict__ q, const unsigned char* __restrict__ mask,
                                float* __restrict__ cmax, float* __restrict__ csum, int Lr) {
  int b = blockIdx.y, col = blockIdx.x, tid = threadIdx.x;
  const float* qb = q + (long)b * Lr * 64;
  const unsigned char* mb = mask + (long)b * Lr;
  __shared__ float red[256];
  float m = -INFINITY;
  for (int l = tid; l < Lr; l += 256)
    if (mb[l]) m = fmaxf(m, qb[(long)l * 64 + col]);
  red[tid] = m; __syncthreads();
  for (int s = 128; s > 0; s >>= 1) { if (tid < s) red[tid] = fmaxf(red[tid], red[tid + s]); __syncthreads(); }
  float cm = red[0];
  __syncthreads();
  float ssum = 0.f;
  if (cm > -INFINITY)
    for (int l = tid; l < Lr; l += 256)
      if (mb[l]) ssum += expf(qb[(long)l * 64 + col] - cm);
  red[tid] = ssum; __syncthreads();
  for (int s = 128; s > 0; s >>= 1) { if (tid < s) red[tid] += red[tid + s]; __syncthreads(); }
  if (tid == 0) { cmax[b * 64 + col] = cm; csum[b * 64 + col] = red[0]; }
}

__global__ void rowsm_kernel(float* __restrict__ q, const unsigned char* __restrict__ mask,
                             const float* __restrict__ cmax, const float* __restrict__ csum, int Lr) {
  long row = blockIdx.x;
  int b = (int)(row / Lr), col = threadIdx.x;
  bool mk = mask[row] != 0;
  float raw = q[row * 64 + col];
  float v = mk ? raw : -INFINITY;
  __shared__ float red[64];
  red[col] = v; __syncthreads();
  for (int s = 32; s > 0; s >>= 1) { if (col < s) red[col] = fmaxf(red[col], red[col + s]); __syncthreads(); }
  float rm = red[0];
  __syncthreads();
  float e = (mk && rm > -INFINITY) ? expf(v - rm) : 0.f;
  red[col] = e; __syncthreads();
  for (int s = 32; s > 0; s >>= 1) { if (col < s) red[col] += red[col + s]; __syncthreads(); }
  float rs = red[0];
  float rsoft = (rs > 0.f) ? e / rs : 0.f;  // nan_to_num(softmax axis=2)
  float cs = csum[b * 64 + col];
  float csoft = 0.f;
  if (mk && cs > 0.f) csoft = expf(raw - cmax[b * 64 + col]) / cs;
  q[row * 64 + col] = csoft * rsoft;
}

// ---------------------------------------------------------------------------
// Main linear attention: K = kproj[B,S,H,8] (f32, post-softmax), V bf16 [B,S,256].
// ---------------------------------------------------------------------------
__global__ void kv_main_kernel(const float* __restrict__ kq, const bf16_t* __restrict__ vb,
                               const unsigned char* __restrict__ smask,
                               float* __restrict__ KV, float* __restrict__ KS, int S) {
  int bh = blockIdx.x, b = bh >> 3, h = bh & 7;
  int t = threadIdx.x, d = t >> 5, v = t & 31;  // 8 x 32 = 256
  __shared__ float sk[8][8], sv[8][32];
  float acc = 0.f, ks = 0.f;
  const float* kb = kq + (long)b * S * 64;
  const bf16_t* vbb = vb + (long)b * S * 256;
  const unsigned char* mb = smask + (long)b * S;
  for (int s0 = 0; s0 < S; s0 += 8) {
    if (t < 64) {
      int i = t >> 3, dd = t & 7;
      float mm = mb[s0 + i] ? 1.f : 0.f;
      sk[i][dd] = elu1(kb[(long)(s0 + i) * 64 + h * 8 + dd]) * mm;
    }
    {
      int i = t >> 5, vv = t & 31;
      float mm = mb[s0 + i] ? 1.f : 0.f;
      sv[i][vv] = (float)vbb[(long)(s0 + i) * 256 + h * 32 + vv] * mm;
    }
    __syncthreads();
#pragma unroll
    for (int i = 0; i < 8; ++i) {
      acc += sk[i][d] * sv[i][v];
      if (v == 0) ks += sk[i][d];
    }
    __syncthreads();
  }
  KV[(long)bh * 256 + d * 32 + v] = acc;
  if (v == 0) KS[bh * 8 + d] = ks;
}

__global__ void attn_apply_main(const float* __restrict__ qf, const float* __restrict__ KV,
                                const float* __restrict__ KS, const unsigned char* __restrict__ qmask,
                                bf16_t* __restrict__ out, int Lr) {
  long row = blockIdx.x;
  int b = (int)(row / Lr);
  int t = threadIdx.x, h = t >> 5, v = t & 31;
  float qm = qmask[row] ? 1.f : 0.f;
  const float* ks = KS + (long)(b * 8 + h) * 8;
  const float* kv = KV + (long)(b * 8 + h) * 256;
  float qe[8], z = 0.f;
#pragma unroll
  for (int d = 0; d < 8; ++d) {
    qe[d] = elu1(qf[row * 64 + h * 8 + d]) * qm;
    z += qe[d] * ks[d];
  }
  z = 1.f / (z + 1e-6f);
  float acc = 0.f;
#pragma unroll
  for (int d = 0; d < 8; ++d) acc += qe[d] * kv[d * 32 + v];
  out[row * 256 + t] = (bf16_t)(acc * z);
}

// ---------------------------------------------------------------------------
// Host orchestration
// ---------------------------------------------------------------------------
extern "C" void kernel_launch(void* const* d_in, const int* in_sizes, int n_in,
                              void* d_out, int out_size, void* d_ws, size_t ws_size,
                              hipStream_t stream) {
  (void)in_sizes; (void)n_in; (void)out_size; (void)ws_size;
  const float* x      = (const float*)d_in[0];
  const float* src    = (const float*)d_in[1];
  const unsigned char* xmask = (const unsigned char*)d_in[2];
  const unsigned char* smask = (const unsigned char*)d_in[3];
  const float* Wq = (const float*)d_in[4],  * Wk = (const float*)d_in[5];
  const float* Wv = (const float*)d_in[6],  * Wqp = (const float*)d_in[7];
  const float* Wkp = (const float*)d_in[8], * Wmerge = (const float*)d_in[9];
  const float* Wmlp1 = (const float*)d_in[10], * Wmlp2 = (const float*)d_in[11];
  const float* ln1g = (const float*)d_in[12], * ln1b = (const float*)d_in[13];
  const float* ln2g = (const float*)d_in[14], * ln2b = (const float*)d_in[15];
  const float* protoq = (const float*)d_in[16];
  const float* dWqq = (const float*)d_in[17], * dWqk = (const float*)d_in[18];
  const float* dWqv = (const float*)d_in[19], * dWqm = (const float*)d_in[20];
  const float* dln1g = (const float*)d_in[21], * dln1b = (const float*)d_in[22];
  const float* dWpq = (const float*)d_in[23], * dWpk = (const float*)d_in[24];
  const float* dWpv = (const float*)d_in[25], * dWpm = (const float*)d_in[26];
  const float* dln2g = (const float*)d_in[27], * dln2b = (const float*)d_in[28];
  const float* dWf1 = (const float*)d_in[29], * dWf2 = (const float*)d_in[30];
  const float* dln3g = (const float*)d_in[31], * dln3b = (const float*)d_in[32];
  float* out = (float*)d_out;

  char* ws = (char*)d_ws;
  size_t off = 0;
  auto alloc = [&](size_t bytes) -> void* {
    void* p = ws + off;
    off += (bytes + 255) & ~(size_t)255;
    return p;
  };

  const size_t NBC2 = (size_t)BL_ * C_ * 2;      // bf16 [B*L,256]
  bf16_t* x_bf  = (bf16_t*)alloc(NBC2);
  bf16_t* s_bf  = (bf16_t*)alloc(NBC2);
  bf16_t* v_bf  = (bf16_t*)alloc(NBC2);
  bf16_t* bufA  = (bf16_t*)alloc(NBC2);          // fk -> q0 -> la_out
  bf16_t* bufB  = (bf16_t*)alloc(NBC2);          // fv -> k0 -> msg_ln
  float*  bigf  = (float*)alloc((size_t)BL_ * C_ * 4);  // merge-out / mlp2-out
  bf16_t* cat   = (bf16_t*)alloc((size_t)BL_ * 512 * 2);
  bf16_t* m1    = (bf16_t*)alloc((size_t)BL_ * 512 * 2);
  // qproj/kproj (f32 [B,L,64]) live in the concat buffer until it is written
  float* qproj = (float*)cat;
  float* kproj = (float*)((char*)cat + (size_t)BL_ * 64 * 4);

  // transposed bf16 weights
  const size_t W2 = (size_t)C_ * C_ * 2;
  bf16_t* wWq = (bf16_t*)alloc(W2),  * wWk = (bf16_t*)alloc(W2),  * wWv = (bf16_t*)alloc(W2);
  bf16_t* wWqp = (bf16_t*)alloc(W2), * wWkp = (bf16_t*)alloc(W2), * wWmg = (bf16_t*)alloc(W2);
  bf16_t* wM1 = (bf16_t*)alloc((size_t)512 * 512 * 2);
  bf16_t* wM2 = (bf16_t*)alloc((size_t)512 * 256 * 2);
  bf16_t* wdqq = (bf16_t*)alloc(W2), * wdqk = (bf16_t*)alloc(W2), * wdqv = (bf16_t*)alloc(W2);
  bf16_t* wdqm = (bf16_t*)alloc(W2), * wdpq = (bf16_t*)alloc(W2), * wdpk = (bf16_t*)alloc(W2);
  bf16_t* wdpv = (bf16_t*)alloc(W2), * wdpm = (bf16_t*)alloc(W2);
  bf16_t* wdf1 = (bf16_t*)alloc(W2), * wdf2 = (bf16_t*)alloc(W2);

  // decoder scratch
  bf16_t* pq_bf = (bf16_t*)alloc((size_t)64 * 256 * 2);
  float* qq_f = (float*)alloc(64 * 256 * 4);
  float* qk_f = (float*)alloc(64 * 256 * 4);
  float* qv_f = (float*)alloc(64 * 256 * 4);
  float* la1_f = (float*)alloc(64 * 256 * 4);
  bf16_t* la1_bf = (bf16_t*)alloc(64 * 256 * 2);
  float* msg0_f = (float*)alloc(64 * 256 * 4);
  float* msg_f = (float*)alloc(64 * 256 * 4);
  bf16_t* msg_bf = (bf16_t*)alloc(64 * 256 * 2);
  float* mq_f = (float*)alloc(64 * 256 * 4);
  float* KV1 = (float*)alloc(8 * 1024 * 4);
  float* KS1 = (float*)alloc(8 * 32 * 4);
  float* KV2 = (float*)alloc(64 * 1024 * 4);
  float* KS2 = (float*)alloc(64 * 32 * 4);
  bf16_t* la2_bf = (bf16_t*)alloc((size_t)B_ * 64 * 256 * 2);
  float* proto1_f = (float*)alloc((size_t)B_ * 64 * 256 * 4);
  float* protoA_f = (float*)alloc((size_t)B_ * 64 * 256 * 4);
  bf16_t* protoA_bf = (bf16_t*)alloc((size_t)B_ * 64 * 256 * 2);
  bf16_t* t1_bf = (bf16_t*)alloc((size_t)B_ * 64 * 256 * 2);
  float* t2_f = (float*)alloc((size_t)B_ * 64 * 256 * 4);
  bf16_t* proto2_bf = (bf16_t*)alloc((size_t)B_ * 64 * 256 * 2);
  bf16_t* pp_bf = (bf16_t*)alloc((size_t)B_ * 64 * 256 * 2);
  bf16_t* pk_bf = (bf16_t*)alloc((size_t)B_ * 64 * 256 * 2);
  float* cmaxq = (float*)alloc(8 * 64 * 4), * csumq = (float*)alloc(8 * 64 * 4);
  float* cmaxk = (float*)alloc(8 * 64 * 4), * csumk = (float*)alloc(8 * 64 * 4);
  float* KVm = (float*)alloc(64 * 256 * 4);
  float* KSm = (float*)alloc(64 * 8 * 4);

  const dim3 gblk(32, 4, 1);  // 4 waves per block

  // -------- conversions --------
  long nxc = (long)BL_ * C_;
  conv_f2b_kernel<<<(unsigned)((nxc + 255) / 256), 256, 0, stream>>>(x, x_bf, nxc);
  conv_f2b_kernel<<<(unsigned)((nxc + 255) / 256), 256, 0, stream>>>(src, s_bf, nxc);
  conv_f2b_kernel<<<64, 256, 0, stream>>>(protoq, pq_bf, 64 * 256);

  const unsigned wgrid = (unsigned)(((long)256 * 256 + 255) / 256);
  wconv_kernel<<<wgrid, 256, 0, stream>>>(Wq, wWq, 256, 256);
  wconv_kernel<<<wgrid, 256, 0, stream>>>(Wk, wWk, 256, 256);
  wconv_kernel<<<wgrid, 256, 0, stream>>>(Wv, wWv, 256, 256);
  wconv_kernel<<<wgrid, 256, 0, stream>>>(Wqp, wWqp, 256, 256);
  wconv_kernel<<<wgrid, 256, 0, stream>>>(Wkp, wWkp, 256, 256);
  wconv_kernel<<<wgrid, 256, 0, stream>>>(Wmerge, wWmg, 256, 256);
  wconv_kernel<<<(512 * 512 + 255) / 256, 256, 0, stream>>>(Wmlp1, wM1, 512, 512);
  wconv_kernel<<<(512 * 256 + 255) / 256, 256, 0, stream>>>(Wmlp2, wM2, 512, 256);
  wconv_kernel<<<wgrid, 256, 0, stream>>>(dWqq, wdqq, 256, 256);
  wconv_kernel<<<wgrid, 256, 0, stream>>>(dWqk, wdqk, 256, 256);
  wconv_kernel<<<wgrid, 256, 0, stream>>>(dWqv, wdqv, 256, 256);
  wconv_kernel<<<wgrid, 256, 0, stream>>>(dWqm, wdqm, 256, 256);
  wconv_kernel<<<wgrid, 256, 0, stream>>>(dWpq, wdpq, 256, 256);
  wconv_kernel<<<wgrid, 256, 0, stream>>>(dWpk, wdpk, 256, 256);
  wconv_kernel<<<wgrid, 256, 0, stream>>>(dWpv, wdpv, 256, 256);
  wconv_kernel<<<wgrid, 256, 0, stream>>>(dWpm, wdpm, 256, 256);
  wconv_kernel<<<wgrid, 256, 0, stream>>>(dWf1, wdf1, 256, 256);
  wconv_kernel<<<wgrid, 256, 0, stream>>>(dWf2, wdf2, 256, 256);

  // -------- decoder front (batch-independent) --------
  gemm_nt<0><<<dim3(8, 1, 1), gblk, 0, stream>>>(pq_bf, wdqq, qq_f, 64, 256, 256, 0, 0, 0);
  gemm_nt<0><<<dim3(8, 1, 1), gblk, 0, stream>>>(pq_bf, wdqk, qk_f, 64, 256, 256, 0, 0, 0);
  gemm_nt<0><<<dim3(8, 1, 1), gblk, 0, stream>>>(pq_bf, wdqv, qv_f, 64, 256, 256, 0, 0, 0);
  kv1_kernel<<<8, 256, 0, stream>>>(qk_f, qv_f, KV1, KS1);
  attn_apply1<<<64, 256, 0, stream>>>(qq_f, KV1, KS1, la1_f);
  conv_f2b_kernel<<<64, 256, 0, stream>>>(la1_f, la1_bf, 64 * 256);
  gemm_nt<0><<<dim3(8, 1, 1), gblk, 0, stream>>>(la1_bf, wdqm, msg0_f, 64, 256, 256, 0, 0, 0);
  ln_kernel<<<64, 256, 0, stream>>>(msg0_f, protoq, 0, nullptr, dln1g, dln1b, msg_f, msg_bf);
  gemm_nt<0><<<dim3(8, 1, 1), gblk, 0, stream>>>(msg_bf, wdpq, mq_f, 64, 256, 256, 0, 0, 0);

  // -------- decoder cross-attention over x --------
  gemm_nt<1><<<dim3(8, 600, 1), gblk, 0, stream>>>(x_bf, wdpk, bufA, BL_, 256, 256, 0, 0, 0);
  gemm_nt<1><<<dim3(8, 600, 1), gblk, 0, stream>>>(x_bf, wdpv, bufB, BL_, 256, 256, 0, 0, 0);
  kv2_kernel<<<64, 256, 0, stream>>>(bufA, bufB, xmask, KV2, KS2, L_);
  attn_apply2<<<dim3(64, 8), 256, 0, stream>>>(mq_f, KV2, KS2, la2_bf);
  gemm_nt<0><<<dim3(8, 1, 8), gblk, 0, stream>>>(la2_bf, wdpm, proto1_f, 64, 256, 256, 16384, 0, 16384);
  ln_kernel<<<512, 256, 0, stream>>>(proto1_f, msg_f, 64, nullptr, dln2g, dln2b, protoA_f, protoA_bf);
  gemm_nt<2><<<dim3(8, 1, 8), gblk, 0, stream>>>(protoA_bf, wdf1, t1_bf, 64, 256, 256, 16384, 0, 16384);
  gemm_nt<0><<<dim3(8, 1, 8), gblk, 0, stream>>>(t1_bf, wdf2, t2_f, 64, 256, 256, 16384, 0, 16384);
  ln_kernel<<<512, 256, 0, stream>>>(t2_f, protoA_f, 0, nullptr, dln3g, dln3b, nullptr, proto2_bf);
  gemm_nt<1><<<dim3(8, 1, 8), gblk, 0, stream>>>(proto2_bf, wWqp, pp_bf, 64, 256, 256, 16384, 0, 16384);
  gemm_nt<1><<<dim3(8, 1, 8), gblk, 0, stream>>>(proto2_bf, wWkp, pk_bf, 64, 256, 256, 16384, 0, 16384);

  // -------- main projections --------
  gemm_nt<1><<<dim3(8, 600, 1), gblk, 0, stream>>>(x_bf, wWq, bufA, BL_, 256, 256, 0, 0, 0);
  gemm_nt<1><<<dim3(8, 600, 1), gblk, 0, stream>>>(s_bf, wWk, bufB, BL_, 256, 256, 0, 0, 0);
  gemm_nt<1><<<dim3(8, 600, 1), gblk, 0, stream>>>(s_bf, wWv, v_bf, BL_, 256, 256, 0, 0, 0);
  gemm_nt<0><<<dim3(2, 75, 8), gblk, 0, stream>>>(bufA, pp_bf, qproj, L_, 64, 256,
                                                  (long)L_ * 256, 16384, (long)L_ * 64);
  gemm_nt<0><<<dim3(2, 75, 8), gblk, 0, stream>>>(bufB, pk_bf, kproj, L_, 64, 256,
                                                  (long)L_ * 256, 16384, (long)L_ * 64);

  // -------- double softmax --------
  colstats_kernel<<<dim3(64, 8), 256, 0, stream>>>(qproj, xmask, cmaxq, csumq, L_);
  rowsm_kernel<<<BL_, 64, 0, stream>>>(qproj, xmask, cmaxq, csumq, L_);
  colstats_kernel<<<dim3(64, 8), 256, 0, stream>>>(kproj, smask, cmaxk, csumk, S_);
  rowsm_kernel<<<BL_, 64, 0, stream>>>(kproj, smask, cmaxk, csumk, S_);

  // -------- main linear attention --------
  kv_main_kernel<<<64, 256, 0, stream>>>(kproj, v_bf, smask, KVm, KSm, S_);
  attn_apply_main<<<BL_, 256, 0, stream>>>(qproj, KVm, KSm, xmask, bufA, L_);

  // -------- merge + MLP --------
  gemm_nt<0><<<dim3(8, 600, 1), gblk, 0, stream>>>(bufA, wWmg, bigf, BL_, 256, 256, 0, 0, 0);
  ln_kernel<<<BL_, 256, 0, stream>>>(bigf, nullptr, 0, nullptr, ln1g, ln1b, nullptr, bufB);
  concat_kernel<<<(unsigned)(((long)BL_ * 512 + 255) / 256), 256, 0, stream>>>(x_bf, bufB, cat, (long)BL_);
  gemm_nt<2><<<dim3(16, 600, 1), gblk, 0, stream>>>(cat, wM1, m1, BL_, 512, 512, 0, 0, 0);
  gemm_nt<0><<<dim3(8, 600, 1), gblk, 0, stream>>>(m1, wM2, bigf, BL_, 256, 512, 0, 0, 0);
  ln_kernel<<<BL_, 256, 0, stream>>>(bigf, nullptr, 0, x, ln2g, ln2b, out, nullptr);
}
